// GTN_58205396795517
// MI455X (gfx1250) — matrix operand order repaired and checked by reading, not compiled, not run on hardware
//
#include <hip/hip_runtime.h>
#include <hip/hip_bf16.h>

typedef __attribute__((ext_vector_type(16))) _Float16 v16h;
typedef __attribute__((ext_vector_type(8)))  float    v8f;

#define NNODES 100000
#define NEDGES 800000
#define DIM    64
#define HEADS  4
#define HC     256   // HEADS * C
#define A_PAD  68    // padded LDS row stride for the A tile (bank-conflict free)

// ---------------------------------------------------------------------------
// CDNA5 async Global->LDS copy (ASYNCcnt path, bypasses VGPRs).
// vdst = LDS byte address, vaddr = 64-bit global address (saddr-null form).
// ---------------------------------------------------------------------------
__device__ __forceinline__ void async_copy_b128(unsigned lds_off, const void* gaddr) {
    asm volatile("global_load_async_to_lds_b128 %0, %1, off"
                 :: "v"(lds_off), "v"((unsigned long long)(uintptr_t)gaddr)
                 : "memory");
}
__device__ __forceinline__ void wait_asynccnt0() {
    asm volatile("s_wait_asynccnt 0x0" ::: "memory");
}

// ---------------------------------------------------------------------------
// Order-preserving float <-> uint mapping so we can use u32 atomicMax for a
// float segment-max (handles negatives correctly).
// ---------------------------------------------------------------------------
__device__ __forceinline__ unsigned f2ord(float f) {
    unsigned u = __float_as_uint(f);
    return (u & 0x80000000u) ? ~u : (u | 0x80000000u);
}
__device__ __forceinline__ float ord2f(unsigned u) {
    return (u & 0x80000000u) ? __uint_as_float(u & 0x7fffffffu)
                             : __uint_as_float(~u);
}

// ---------------------------------------------------------------------------
// WMMA GEMM with async-LDS staging:
//   C[M x ncols] = A[M x 64] * B[64 x ncols] + bias[ncols]
// One block (8 waves) per 16-row m-block:
//   - B panel (64 x ncols f32) async-staged to LDS once per block
//   - A tile (16 x 64 f32) async-staged to padded LDS once per block
//   - each wave keeps its A fragments in registers and sweeps n-tiles
// Fragment layouts per CDNA5 ISA 7.12.2:
//   A 16x32 f16: lanes 0-15 row=lane, elems 0..7 -> K kb..kb+7, 8..15 -> K kb+16..;
//                kb = 0 (lanes<16) / 8 (lanes>=16)
//   B 32x16 f16: col=lane&15, same K order as A (elems 0..7 -> kb..kb+7, 8..15 -> kb+16..)
//   C/D 16x16 f32: VGPR r: lanes 0-15 M=r, lanes 16-31 M=r+8; N = lane&15
// ---------------------------------------------------------------------------
__global__ __launch_bounds__(256)
void gemm_wmma_lds(const float* __restrict__ A, const float* __restrict__ B,
                   const float* __restrict__ bias, float* __restrict__ C,
                   int ncols)
{
    extern __shared__ float smem[];
    float* sB = smem;                         // [64][ncols]
    float* sA = smem + 64 * ncols;            // [16][A_PAD]

    const int tid  = threadIdx.x;
    const int lane = tid & 31;
    const int wave = tid >> 5;
    const int m0   = blockIdx.x * 16;

    // ---- async-stage B panel: 64*ncols floats = 16*ncols float4 transfers ----
    const int nvecB = 16 * ncols;
    for (int i = tid; i < nvecB; i += 256) {
        async_copy_b128((unsigned)(uintptr_t)(sB + (size_t)i * 4),
                        B + (size_t)i * 4);
    }
    // ---- async-stage A tile: 16 rows x 16 float4 (padded row stride) ----
    {
        const int row = tid >> 4;             // 0..15
        const int c4  = tid & 15;             // 0..15
        async_copy_b128((unsigned)(uintptr_t)(sA + row * A_PAD + c4 * 4),
                        A + (size_t)(m0 + row) * DIM + c4 * 4);
    }
    wait_asynccnt0();
    __syncthreads();

    // ---- A fragments in registers, reused across all n-tiles of this wave ----
    const int akb = (lane < 16) ? 0 : 8;
    const int bkb = (lane < 16) ? 0 : 8;
    const float* sArow = sA + (lane & 15) * A_PAD;
    v16h a0, a1;
    #pragma unroll
    for (int i = 0; i < 8; ++i) {
        a0[i]     = (_Float16)sArow[akb + i];
        a0[i + 8] = (_Float16)sArow[akb + 16 + i];
        a1[i]     = (_Float16)sArow[32 + akb + i];
        a1[i + 8] = (_Float16)sArow[32 + akb + 16 + i];
    }

    const int tiles_n = ncols >> 4;
    for (int t = wave; t < tiles_n; t += 8) {         // wave-uniform: EXEC all-1s
        const int col = t * 16 + (lane & 15);

        v8f acc;
        {
            const float bv = bias[col];               // bias depends only on N
            #pragma unroll
            for (int r = 0; r < 8; ++r) acc[r] = bv;
        }

        v16h b0, b1;
        #pragma unroll
        for (int i = 0; i < 16; ++i) {
            b0[i] = (_Float16)sB[(size_t)(bkb + (i & 7) + ((i & 8) << 1)) * ncols + col];
            b1[i] = (_Float16)sB[(size_t)(32 + bkb + (i & 7) + ((i & 8) << 1)) * ncols + col];
        }
        acc = __builtin_amdgcn_wmma_f32_16x16x32_f16(false, a0, false, b0,
                                                     (short)0, acc, false, false);
        acc = __builtin_amdgcn_wmma_f32_16x16x32_f16(false, a1, false, b1,
                                                     (short)0, acc, false, false);

        const int rbase = m0 + ((lane < 16) ? 0 : 8);
        #pragma unroll
        for (int r = 0; r < 8; ++r) {
            C[(size_t)(rbase + r) * ncols + col] = acc[r];
        }
    }
}

// ---------------------------------------------------------------------------
// Reset per-layer accumulators: agg=0, denom=0, amax=-inf (ordered space)
// ---------------------------------------------------------------------------
__global__ __launch_bounds__(256)
void init_kernel(unsigned* __restrict__ amax, float* __restrict__ denom,
                 float* __restrict__ agg)
{
    const long long t = (long long)blockIdx.x * 256 + threadIdx.x;
    if (t < (long long)NNODES * HC) agg[t] = 0.0f;
    if (t < (long long)NNODES * HEADS) {
        denom[t] = 0.0f;
        amax[t]  = f2ord(-INFINITY);
    }
}

// ---------------------------------------------------------------------------
// Per-edge attention logits + running segment max.
// One wave per edge; lane handles 8 contiguous channels -> head = lane/8.
// ---------------------------------------------------------------------------
__global__ __launch_bounds__(256)
void edge_alpha_kernel(const int* __restrict__ ei, const float* __restrict__ q,
                       const float* __restrict__ k, float* __restrict__ alpha,
                       unsigned* __restrict__ amax)
{
    const int lane = threadIdx.x & 31;
    const int e    = (int)((blockIdx.x * blockDim.x + threadIdx.x) >> 5);
    if (e >= NEDGES) return;
    const int src = ei[e];
    const int dst = ei[NEDGES + e];

    const float* qd = q + (size_t)dst * HC + lane * 8;
    const float* ks = k + (size_t)src * HC + lane * 8;
    float s = 0.0f;
    #pragma unroll
    for (int j = 0; j < 8; ++j) s += qd[j] * ks[j];

    // reduce within each 8-lane head group (wave32)
    s += __shfl_xor(s, 1, 32);
    s += __shfl_xor(s, 2, 32);
    s += __shfl_xor(s, 4, 32);

    if ((lane & 7) == 0) {
        const int   h = lane >> 3;
        const float a = s * 0.125f;           // 1/sqrt(64)
        alpha[(size_t)e * HEADS + h] = a;
        atomicMax(&amax[(size_t)dst * HEADS + h], f2ord(a));
    }
}

// ---------------------------------------------------------------------------
// ex = exp(alpha - amax[dst]); denom[dst] += ex   (one thread per (e,h))
// ---------------------------------------------------------------------------
__global__ __launch_bounds__(256)
void edge_exp_kernel(const int* __restrict__ ei, float* __restrict__ alpha,
                     const unsigned* __restrict__ amax, float* __restrict__ denom)
{
    const long long t = (long long)blockIdx.x * 256 + threadIdx.x;
    if (t >= (long long)NEDGES * HEADS) return;
    const int e   = (int)(t >> 2);
    const int h   = (int)(t & 3);
    const int dst = ei[NEDGES + e];
    const float m  = ord2f(amax[(size_t)dst * HEADS + h]);
    const float ex = __expf(alpha[t] - m);
    alpha[t] = ex;                                      // reuse buffer for ex
    atomicAdd(&denom[(size_t)dst * HEADS + h], ex);
}

// ---------------------------------------------------------------------------
// agg[dst,h,c] += v[src,h,c] * attn[e,h]   (one thread per (e,c), 4 heads)
// ---------------------------------------------------------------------------
__global__ __launch_bounds__(256)
void edge_agg_kernel(const int* __restrict__ ei, const float* __restrict__ v,
                     const float* __restrict__ ex, const float* __restrict__ denom,
                     float* __restrict__ agg)
{
    const long long t = (long long)blockIdx.x * 256 + threadIdx.x;
    if (t >= (long long)NEDGES * DIM) return;
    const int e   = (int)(t >> 6);
    const int c   = (int)(t & 63);
    const int src = ei[e];
    const int dst = ei[NEDGES + e];
    #pragma unroll
    for (int h = 0; h < HEADS; ++h) {
        const float attn = ex[(size_t)e * HEADS + h]
                         / (denom[(size_t)dst * HEADS + h] + 1e-16f);
        const float val  = v[(size_t)src * HC + h * DIM + c] * attn;
        atomicAdd(&agg[(size_t)dst * HC + h * DIM + c], val);
    }
}

// ---------------------------------------------------------------------------
// out[n,c] = mean_h(agg[n,h,c]) + skip[n,c]  (optional ReLU)
// ---------------------------------------------------------------------------
__global__ __launch_bounds__(256)
void finalize_kernel(const float* __restrict__ agg, const float* __restrict__ skip,
                     float* __restrict__ out, int relu)
{
    const long long t = (long long)blockIdx.x * 256 + threadIdx.x;
    if (t >= (long long)NNODES * DIM) return;
    const int n = (int)(t >> 6);
    const int c = (int)(t & 63);
    const size_t base = (size_t)n * HC + c;
    float r = 0.25f * (agg[base] + agg[base + 64] + agg[base + 128] + agg[base + 192])
            + skip[t];
    out[t] = relu ? fmaxf(r, 0.0f) : r;
}

// ---------------------------------------------------------------------------
// Host-side layer driver
// ---------------------------------------------------------------------------
static void run_layer(const float* xin, const int* ei,
                      const float* Wq, const float* bq,
                      const float* Wk, const float* bk,
                      const float* Wv, const float* bv,
                      const float* Ws, const float* bs,
                      float* q, float* k, float* v, float* s,
                      float* alpha, float* denom, float* agg, unsigned* amax,
                      float* yout, int relu, hipStream_t stream)
{
    const int    mblocks    = NNODES / 16;                       // 6250
    const size_t shmem_qkv  = (size_t)(64 * HC  + 16 * A_PAD) * sizeof(float);
    const size_t shmem_skip = (size_t)(64 * DIM + 16 * A_PAD) * sizeof(float);

    gemm_wmma_lds<<<mblocks, 256, shmem_qkv,  stream>>>(xin, Wq, bq, q, HC);
    gemm_wmma_lds<<<mblocks, 256, shmem_qkv,  stream>>>(xin, Wk, bk, k, HC);
    gemm_wmma_lds<<<mblocks, 256, shmem_qkv,  stream>>>(xin, Wv, bv, v, HC);
    gemm_wmma_lds<<<mblocks, 256, shmem_skip, stream>>>(xin, Ws, bs, s, DIM);

    const int init_blocks  = (int)(((long long)NNODES * HC + 255) / 256);
    init_kernel<<<init_blocks, 256, 0, stream>>>(amax, denom, agg);

    const int alpha_blocks = (NEDGES * 32 + 255) / 256;
    edge_alpha_kernel<<<alpha_blocks, 256, 0, stream>>>(ei, q, k, alpha, amax);

    const int exp_blocks = (int)(((long long)NEDGES * HEADS + 255) / 256);
    edge_exp_kernel<<<exp_blocks, 256, 0, stream>>>(ei, alpha, amax, denom);

    const int agg_blocks = (int)(((long long)NEDGES * DIM + 255) / 256);
    edge_agg_kernel<<<agg_blocks, 256, 0, stream>>>(ei, v, alpha, denom, agg);

    const int fin_blocks = (int)(((long long)NNODES * DIM + 255) / 256);
    finalize_kernel<<<fin_blocks, 256, 0, stream>>>(agg, s, yout, relu);
}

extern "C" void kernel_launch(void* const* d_in, const int* in_sizes, int n_in,
                              void* d_out, int out_size, void* d_ws, size_t ws_size,
                              hipStream_t stream)
{
    const float* x  = (const float*)d_in[0];
    const int*   ei = (const int*)d_in[1];
    const float* Wq1 = (const float*)d_in[2];  const float* bq1 = (const float*)d_in[3];
    const float* Wk1 = (const float*)d_in[4];  const float* bk1 = (const float*)d_in[5];
    const float* Wv1 = (const float*)d_in[6];  const float* bv1 = (const float*)d_in[7];
    const float* Ws1 = (const float*)d_in[8];  const float* bs1 = (const float*)d_in[9];
    const float* Wq2 = (const float*)d_in[10]; const float* bq2 = (const float*)d_in[11];
    const float* Wk2 = (const float*)d_in[12]; const float* bk2 = (const float*)d_in[13];
    const float* Wv2 = (const float*)d_in[14]; const float* bv2 = (const float*)d_in[15];
    const float* Ws2 = (const float*)d_in[16]; const float* bs2 = (const float*)d_in[17];

    // ---- workspace layout (floats) ----
    float* ws = (float*)d_ws;
    float* q     = ws;                              // N*256
    float* k     = q     + (size_t)NNODES * HC;     // N*256
    float* v     = k     + (size_t)NNODES * HC;     // N*256
    float* s     = v     + (size_t)NNODES * HC;     // N*64 (skip GEMM out)
    float* hbuf  = s     + (size_t)NNODES * DIM;    // N*64 (layer-1 output)
    float* alpha = hbuf  + (size_t)NNODES * DIM;    // E*4  (logits -> exp)
    float* denom = alpha + (size_t)NEDGES * HEADS;  // N*4
    float* agg   = denom + (size_t)NNODES * HEADS;  // N*256
    unsigned* amax = (unsigned*)(agg + (size_t)NNODES * HC);  // N*4

    float* out = (float*)d_out;

    // Layer 1: x -> hbuf (with ReLU)
    run_layer(x, ei, Wq1, bq1, Wk1, bk1, Wv1, bv1, Ws1, bs1,
              q, k, v, s, alpha, denom, agg, amax, hbuf, /*relu=*/1, stream);

    // Layer 2: hbuf -> out (no ReLU)
    run_layer(hbuf, ei, Wq2, bq2, Wk2, bk2, Wv2, bv2, Ws2, bs2,
              q, k, v, s, alpha, denom, agg, amax, out, /*relu=*/0, stream);
}
